// Qwen2MoTDecoderLayer_16183436771514
// MI455X (gfx1250) — matrix operand order repaired and checked by reading, not compiled, not run on hardware
//
#include <hip/hip_runtime.h>
#include <hip/hip_bf16.h>

#define T_TOK 8192
#define B_SZ  8
#define L_SEQ 1024
#define HIDD  2048
#define NH    16
#define NKV   8
#define HD    128
#define EPSF  1e-6f
#define SCALE 0.08838834764831845f   // 1/sqrt(128)

typedef __attribute__((ext_vector_type(16))) __bf16 v16bf;
typedef __attribute__((ext_vector_type(8)))  __bf16 v8bf;
typedef __attribute__((ext_vector_type(8)))  float  v8f;

__device__ inline v16bf combine16(v8bf lo, v8bf hi) {
  return __builtin_shufflevector(lo, hi, 0, 1, 2, 3, 4, 5, 6, 7,
                                 8, 9, 10, 11, 12, 13, 14, 15);
}

// ---------------- WMMA fragment loaders (bf16, 16x16x32) -------------------
// A operand, 16x32 (MxK): lane 0-15 -> row=lane, K in {kh..kh+7, kh+16..kh+23},
// kh = 0 for lanes 0-15, 8 for lanes 16-31 (ISA 7.12.2). Two aligned 16B loads.
__device__ inline v16bf load_a_frag(const __bf16* __restrict__ rowptr, int lane) {
  const int kh = (lane >> 4) << 3;       // 0 or 8
  v8bf lo = *(const v8bf*)(rowptr + kh);
  v8bf hi = *(const v8bf*)(rowptr + kh + 16);
  return combine16(lo, hi);
}
// B operand, 32x16 (KxN): lane n (0-15) holds column n, K in {kh..kh+7, kh+16..kh+23},
// kh = 0 for lanes 0-15, 8 for lanes 16-31 (same as A). Two aligned 16B loads.
__device__ inline v16bf load_b_frag(const __bf16* __restrict__ colptr, int lane) {
  const int kh = (lane >> 4) << 3;       // 0 or 8
  v8bf lo = *(const v8bf*)(colptr + kh);
  v8bf hi = *(const v8bf*)(colptr + kh + 16);
  return combine16(lo, hi);
}

__device__ inline v8f vzero8() {
  v8f v;
#pragma unroll
  for (int i = 0; i < 8; ++i) v[i] = 0.0f;
  return v;
}

// ---------------- f32 -> bf16 convert --------------------------------------
__global__ void cvt_bf16_kernel(const float* __restrict__ in,
                                __bf16* __restrict__ out, int n) {
  int i = blockIdx.x * blockDim.x + threadIdx.x;
  if (i < n) out[i] = (__bf16)in[i];
}

// ---------------- routed GEMM: out[2m+r, n] = x[2m+r,:] . W_r[n,:] + b_r[n] -
// Block: 128 threads = 4 waves; tile 16(M) x 64(N); K looped in steps of 32.
__global__ __launch_bounds__(128)
void routed_gemm_kernel(const __bf16* __restrict__ A,
                        const __bf16* __restrict__ Wu,
                        const __bf16* __restrict__ Wg,
                        const float* __restrict__ bu,
                        const float* __restrict__ bg,
                        float* __restrict__ out, int K, int ldout)
{
  const int lane = threadIdx.x & 31;
  const int wave = threadIdx.x >> 5;
  const int r  = blockIdx.z;
  const int m0 = blockIdx.x * 16;
  const int n0 = blockIdx.y * 64 + wave * 16;
  const __bf16* W   = r ? Wg : Wu;
  const float* bias = r ? bg : bu;
  const int lr = lane & 15;
  const size_t arow = (size_t)(2 * (m0 + lr) + r) * K;
  const size_t wrow = (size_t)(n0 + lr) * K;

  v8f acc = vzero8();
  for (int k0 = 0; k0 < K; k0 += 32) {
    // keep the L2 -> WGP pipe primed three tiles ahead (global_prefetch_b8)
    if (k0 + 96 < K) {
      __builtin_prefetch(A + arow + k0 + 96, 0, 0);
      __builtin_prefetch(W + wrow + k0 + 96, 0, 0);
    }
    v16bf a = load_a_frag(A + arow + k0, lane);
    v16bf b = load_b_frag(W + wrow + k0, lane);
    acc = __builtin_amdgcn_wmma_f32_16x16x32_bf16(false, a, false, b,
                                                  (short)0, acc, false, false);
  }
  const int col   = n0 + lr;
  const float bv  = bias ? bias[col] : 0.0f;
  const int rbase = (lane >> 4) << 3;
#pragma unroll
  for (int i = 0; i < 8; ++i) {
    const int t = 2 * (m0 + rbase + i) + r;
    out[(size_t)t * ldout + col] = acc[i] + bv;
  }
}

// ---------------- per-(token, head) RMSNorm + RoPE, f32 in -> bf16 out ------
__global__ __launch_bounds__(128)
void rms_rope_kernel(const float* __restrict__ in,
                     const float* __restrict__ wu, const float* __restrict__ wg,
                     const float* __restrict__ cs, const float* __restrict__ sn,
                     __bf16* __restrict__ out, int heads)
{
  __shared__ float sm[128];
  const int t = blockIdx.x, h = blockIdx.y, d = threadIdx.x;
  const size_t idx = ((size_t)t * heads + h) * HD + d;
  const float x = in[idx];
  sm[d] = x * x;
  __syncthreads();
#pragma unroll
  for (int s = 64; s > 0; s >>= 1) {
    if (d < s) sm[d] += sm[d + s];
    __syncthreads();
  }
  const float inv = rsqrtf(sm[0] * (1.0f / HD) + EPSF);
  __syncthreads();
  const float w  = (t & 1) ? wg[d] : wu[d];
  const float xn = x * inv * w;
  sm[d] = xn;
  __syncthreads();
  const float rot = (d < 64) ? -sm[d + 64] : sm[d - 64];
  const float v = xn * cs[(size_t)t * HD + d] + rot * sn[(size_t)t * HD + d];
  out[idx] = (__bf16)v;
}

// ---------------- V transpose: (T,KV,D) f32 -> [B][KV][D][L] bf16 ----------
__global__ void transpose_v_kernel(const float* __restrict__ vf,
                                   __bf16* __restrict__ vt) {
  int i = blockIdx.x * blockDim.x + threadIdx.x;    // over T*KV*D
  if (i >= T_TOK * NKV * HD) return;
  const int d  = i % HD;
  const int kv = (i / HD) % NKV;
  const int t  = i / (HD * NKV);
  const int b  = t / L_SEQ, l = t % L_SEQ;
  vt[((size_t)(b * NKV + kv) * HD + d) * L_SEQ + l] = (__bf16)vf[i];
}

// ---------------- flash attention: one wave per (b, h, 16-row q tile) ------
__global__ __launch_bounds__(32)
void flash_attn_kernel(const __bf16* __restrict__ qb,
                       const __bf16* __restrict__ kb,
                       const __bf16* __restrict__ vt,
                       __bf16* __restrict__ ob)
{
  __shared__ __align__(16) __bf16 pshm[16 * 32];
  const int lane = threadIdx.x;
  const int q0 = blockIdx.x * 16;
  const int h  = blockIdx.y;
  const int b  = blockIdx.z;
  const int kv = h >> 1;                 // G = NH/NKV = 2
  const int lr = lane & 15;
  const int half = lane >> 4;
  const int rbase = half << 3;

  // Q tile: 16 rows x 128 d, as 4 A-frags over d
  v16bf aq[4];
#pragma unroll
  for (int dc = 0; dc < 4; ++dc)
    aq[dc] = load_a_frag(
        qb + ((size_t)(b * L_SEQ + q0 + lr) * NH + h) * HD + dc * 32, lane);

  v8f o[8];
#pragma unroll
  for (int f = 0; f < 8; ++f) o[f] = vzero8();
  float m_i[8], l_i[8];
#pragma unroll
  for (int i = 0; i < 8; ++i) { m_i[i] = -3.0e38f; l_i[i] = 0.0f; }

  for (int key0 = 0; key0 <= q0 + 15; key0 += 32) {
    // ---- scores: S(16x32) = Q(16x128) . K^T, two 16x16 C-frags ----
    v8f s0 = vzero8(), s1 = vzero8();
#pragma unroll
    for (int dc = 0; dc < 4; ++dc) {
      v16bf bk0 = load_b_frag(
          kb + ((size_t)(b * L_SEQ + key0 + lr) * NKV + kv) * HD + dc * 32, lane);
      s0 = __builtin_amdgcn_wmma_f32_16x16x32_bf16(false, aq[dc], false, bk0,
                                                   (short)0, s0, false, false);
      v16bf bk1 = load_b_frag(
          kb + ((size_t)(b * L_SEQ + key0 + 16 + lr) * NKV + kv) * HD + dc * 32, lane);
      s1 = __builtin_amdgcn_wmma_f32_16x16x32_bf16(false, aq[dc], false, bk1,
                                                   (short)0, s1, false, false);
    }
    // ---- scale + causal mask (-1e9 like the reference) ----
#pragma unroll
    for (int i = 0; i < 8; ++i) {
      const int row = q0 + rbase + i;
      const int c0 = key0 + lr, c1 = c0 + 16;
      s0[i] = s0[i] * SCALE + (c0 <= row ? 0.0f : -1.0e9f);
      s1[i] = s1[i] * SCALE + (c1 <= row ? 0.0f : -1.0e9f);
    }
    // ---- online softmax (row reductions within 16-lane halves) ----
#pragma unroll
    for (int i = 0; i < 8; ++i) {
      float mx = fmaxf(s0[i], s1[i]);
      mx = fmaxf(mx, __shfl_xor(mx, 1, 32));
      mx = fmaxf(mx, __shfl_xor(mx, 2, 32));
      mx = fmaxf(mx, __shfl_xor(mx, 4, 32));
      mx = fmaxf(mx, __shfl_xor(mx, 8, 32));
      const float mnew = fmaxf(m_i[i], mx);
      const float corr = __expf(m_i[i] - mnew);
      m_i[i] = mnew;
      const float p0 = __expf(s0[i] - mnew);
      const float p1 = __expf(s1[i] - mnew);
      s0[i] = p0; s1[i] = p1;
      float rs = p0 + p1;
      rs += __shfl_xor(rs, 1, 32);
      rs += __shfl_xor(rs, 2, 32);
      rs += __shfl_xor(rs, 4, 32);
      rs += __shfl_xor(rs, 8, 32);
      l_i[i] = l_i[i] * corr + rs;
#pragma unroll
      for (int f = 0; f < 8; ++f) o[f][i] *= corr;
    }
    // ---- C-layout P -> A-layout P via LDS ----
#pragma unroll
    for (int i = 0; i < 8; ++i) {
      pshm[(rbase + i) * 32 + lr]      = (__bf16)s0[i];
      pshm[(rbase + i) * 32 + lr + 16] = (__bf16)s1[i];
    }
    __syncthreads();
    v16bf pa;
    {
      const int kh = half << 3;
      v8bf lo = *(const v8bf*)(&pshm[lr * 32 + kh]);
      v8bf hi = *(const v8bf*)(&pshm[lr * 32 + kh + 16]);
      pa = combine16(lo, hi);
    }
    // ---- O += P(16x32) . V(32x16 per d-chunk) ----
#pragma unroll
    for (int f = 0; f < 8; ++f) {
      v16bf bvf = load_b_frag(
          vt + ((size_t)(b * NKV + kv) * HD + f * 16 + lr) * L_SEQ + key0, lane);
      o[f] = __builtin_amdgcn_wmma_f32_16x16x32_bf16(false, pa, false, bvf,
                                                     (short)0, o[f], false, false);
    }
    __syncthreads();
  }

  // ---- normalize and store o (bf16, layout (T, H*D)) ----
#pragma unroll
  for (int i = 0; i < 8; ++i) {
    const float inv = 1.0f / l_i[i];
    const size_t trow =
        (size_t)(b * L_SEQ + q0 + rbase + i) * (NH * HD) + h * HD;
#pragma unroll
    for (int f = 0; f < 8; ++f)
      ob[trow + f * 16 + lr] = (__bf16)(o[f][i] * inv);
  }
}

// ---------------------------------------------------------------------------
extern "C" void kernel_launch(void* const* d_in, const int* in_sizes, int n_in,
                              void* d_out, int out_size, void* d_ws, size_t ws_size,
                              hipStream_t stream) {
  (void)in_sizes; (void)n_in; (void)out_size; (void)ws_size;
  const float* x    = (const float*)d_in[0];
  const float* cosT = (const float*)d_in[1];
  const float* sinT = (const float*)d_in[2];
  // d_in[3] attention_mask (causal, applied analytically); d_in[4/5] idx = even/odd
  const float* Wq_u = (const float*)d_in[6];
  const float* bq_u = (const float*)d_in[7];
  const float* Wk_u = (const float*)d_in[8];
  const float* bk_u = (const float*)d_in[9];
  const float* Wv_u = (const float*)d_in[10];
  const float* bv_u = (const float*)d_in[11];
  const float* Wo_u = (const float*)d_in[12];
  const float* Wq_g = (const float*)d_in[13];
  const float* bq_g = (const float*)d_in[14];
  const float* Wk_g = (const float*)d_in[15];
  const float* bk_g = (const float*)d_in[16];
  const float* Wv_g = (const float*)d_in[17];
  const float* bv_g = (const float*)d_in[18];
  const float* Wo_g = (const float*)d_in[19];
  const float* qn_u = (const float*)d_in[20];
  const float* kn_u = (const float*)d_in[21];
  const float* qn_g = (const float*)d_in[22];
  const float* kn_g = (const float*)d_in[23];

  char* ws = (char*)d_ws;
  size_t off = 0;
  auto alloc = [&](size_t bytes) -> void* {
    void* p = ws + off;
    off += (bytes + 255) & ~(size_t)255;
    return p;
  };
  __bf16* xb   = (__bf16*)alloc((size_t)T_TOK * HIDD * 2);
  __bf16* wqub = (__bf16*)alloc((size_t)HIDD * HIDD * 2);
  __bf16* wqgb = (__bf16*)alloc((size_t)HIDD * HIDD * 2);
  __bf16* wkub = (__bf16*)alloc((size_t)NKV * HD * HIDD * 2);
  __bf16* wkgb = (__bf16*)alloc((size_t)NKV * HD * HIDD * 2);
  __bf16* wvub = (__bf16*)alloc((size_t)NKV * HD * HIDD * 2);
  __bf16* wvgb = (__bf16*)alloc((size_t)NKV * HD * HIDD * 2);
  __bf16* woub = (__bf16*)alloc((size_t)HIDD * HIDD * 2);
  __bf16* wogb = (__bf16*)alloc((size_t)HIDD * HIDD * 2);
  float*  qf   = (float*)alloc((size_t)T_TOK * NH * HD * 4);
  float*  kf   = (float*)alloc((size_t)T_TOK * NKV * HD * 4);
  float*  vf   = (float*)alloc((size_t)T_TOK * NKV * HD * 4);
  __bf16* qb   = (__bf16*)alloc((size_t)T_TOK * NH * HD * 2);
  __bf16* kb   = (__bf16*)alloc((size_t)T_TOK * NKV * HD * 2);
  __bf16* vt   = (__bf16*)alloc((size_t)T_TOK * NKV * HD * 2);
  __bf16* ob   = (__bf16*)alloc((size_t)T_TOK * NH * HD * 2);

  const int CT = 256;
  auto cvt = [&](const float* src, __bf16* dst, int n) {
    cvt_bf16_kernel<<<(n + CT - 1) / CT, CT, 0, stream>>>(src, dst, n);
  };
  cvt(x,    xb,   T_TOK * HIDD);
  cvt(Wq_u, wqub, HIDD * HIDD);
  cvt(Wq_g, wqgb, HIDD * HIDD);
  cvt(Wk_u, wkub, NKV * HD * HIDD);
  cvt(Wk_g, wkgb, NKV * HD * HIDD);
  cvt(Wv_u, wvub, NKV * HD * HIDD);
  cvt(Wv_g, wvgb, NKV * HD * HIDD);
  cvt(Wo_u, woub, HIDD * HIDD);
  cvt(Wo_g, wogb, HIDD * HIDD);

  // QKV projections (routed)
  routed_gemm_kernel<<<dim3(T_TOK / 32, HIDD / 64, 2), 128, 0, stream>>>(
      xb, wqub, wqgb, bq_u, bq_g, qf, HIDD, NH * HD);
  routed_gemm_kernel<<<dim3(T_TOK / 32, (NKV * HD) / 64, 2), 128, 0, stream>>>(
      xb, wkub, wkgb, bk_u, bk_g, kf, HIDD, NKV * HD);
  routed_gemm_kernel<<<dim3(T_TOK / 32, (NKV * HD) / 64, 2), 128, 0, stream>>>(
      xb, wvub, wvgb, bv_u, bv_g, vf, HIDD, NKV * HD);

  // RMSNorm + RoPE (routed weights by token parity)
  rms_rope_kernel<<<dim3(T_TOK, NH), 128, 0, stream>>>(
      qf, qn_u, qn_g, cosT, sinT, qb, NH);
  rms_rope_kernel<<<dim3(T_TOK, NKV), 128, 0, stream>>>(
      kf, kn_u, kn_g, cosT, sinT, kb, NKV);

  // V: f32 (T,KV,D) -> bf16 [B][KV][D][L]
  transpose_v_kernel<<<(T_TOK * NKV * HD + CT - 1) / CT, CT, 0, stream>>>(vf, vt);

  // Flash attention: one wave per (b, h, 16-row q tile)
  flash_attn_kernel<<<dim3(L_SEQ / 16, NH, B_SZ), 32, 0, stream>>>(qb, kb, vt, ob);

  // Output projection (routed), f32 result straight into d_out
  routed_gemm_kernel<<<dim3(T_TOK / 32, HIDD / 64, 2), 128, 0, stream>>>(
      ob, woub, wogb, nullptr, nullptr, (float*)d_out, NH * HD, HIDD);
}
